// multiStepModel_66907000537377
// MI455X (gfx1250) — matrix operand order repaired and checked by reading, not compiled, not run on hardware
//
#include <hip/hip_runtime.h>
#include <hip/hip_bf16.h>

#define DEVINL __device__ __forceinline__

typedef _Float16 v16h __attribute__((ext_vector_type(16)));
typedef _Float16 v8h  __attribute__((ext_vector_type(8)));
typedef float    v8f  __attribute__((ext_vector_type(8)));

// ---- model dims ----
#define BB    128
#define TT    365
#define INW   64
#define COV   256
#define H1    512
#define H2    512
#define OLEN  90
#define NWG   16      // == max workgroups per CDNA5 cluster
#define NTHR  256     // 8 wave32 per WG

// padded LDS row strides (halfs) -> conflict-free ds_load_b128, 16B aligned rows
#define LDSTR1 584    // encoder: K=576 (+8 pad); 292 dwords, rl*36 mod 64 disjoint
#define LDSTR2 520    // decoder: K=512 (+8 pad); 260 dwords, rl*4  mod 64 disjoint
#define LDS_BYTES (128 * LDSTR1 * 2)   // 149504 B dynamic LDS (<< 320KB WGP limit)

union HV16 { v16h v; v8h h[2]; };

DEVINL v8f wmma32f16(v16h a, v16h b, v8f c) {
  // v_wmma_f32_16x16x32_f16 : D = A(16x32) x B(32x16) + C(16x16 f32)
  return __builtin_amdgcn_wmma_f32_16x16x32_f16(false, a, false, b, (short)0, c, false, false);
}

// A fragment (M=16 x K=32), 16-bit, source row-major [row][k], leading dim ld.
// Layout: lanes 0-15 rows M=0..15 hold K {0..15}; lanes 16-31 hold K {16..31}.
DEVINL v16h load_afrag(const _Float16* __restrict__ base, int ld, int m0, int k0, int lane) {
  const int hf = lane >> 4, rr = lane & 15;
  const _Float16* p = base + (size_t)(m0 + rr) * ld + k0 + hf * 16;
  HV16 a;
  a.h[0] = *(const v8h*)(p);        // K = k0 + hf*16     .. +7
  a.h[1] = *(const v8h*)(p + 8);    // K = k0 + hf*16 + 8 .. +15
  return a.v;
}

// B fragment (K=32 x N=16) from GLOBAL weights stored [g][k] row-major.
DEVINL v16h load_bfrag_g(const _Float16* __restrict__ wT, int ldk, int gcol, int k0, int lane) {
  const int hf = lane >> 4;
  const _Float16* p = wT + (size_t)gcol * ldk + k0 + hf * 16;
  HV16 b;
  b.h[0] = *(const v8h*)(p);
  b.h[1] = *(const v8h*)(p + 8);
  return b.v;
}

// B fragment from the LDS-staged weight slice: rows = 128 local gate cols.
DEVINL v16h load_bfrag_lds(const _Float16* lds, int stride, int gl, int k0, int hf) {
  const _Float16* p = lds + (size_t)gl * stride + k0 + hf * 16;
  HV16 b;
  b.h[0] = *(const v8h*)(p);        // ds_load_b128
  b.h[1] = *(const v8h*)(p + 8);    // ds_load_b128
  return b.v;
}

DEVINL float sigm(float x) { return 1.0f / (1.0f + __expf(-x)); }

// Device-wide barrier: atomic generation counter (functional under plain dispatch)
// plus the CDNA5 cluster split-barrier (NOP when not cluster-dispatched).
DEVINL void grid_sync(unsigned* bar) {
  __threadfence();
  __syncthreads();
  if (threadIdx.x == 0) {
    unsigned gen = __hip_atomic_load(bar + 1, __ATOMIC_RELAXED, __HIP_MEMORY_SCOPE_AGENT);
    unsigned arr = __hip_atomic_fetch_add(bar + 0, 1u, __ATOMIC_ACQ_REL, __HIP_MEMORY_SCOPE_AGENT);
    if (arr == (unsigned)(NWG - 1)) {
      __hip_atomic_store(bar + 0, 0u, __ATOMIC_RELAXED, __HIP_MEMORY_SCOPE_AGENT);
      __hip_atomic_fetch_add(bar + 1, 1u, __ATOMIC_RELEASE, __HIP_MEMORY_SCOPE_AGENT);
    } else {
      while (__hip_atomic_load(bar + 1, __ATOMIC_ACQUIRE, __HIP_MEMORY_SCOPE_AGENT) == gen)
        __builtin_amdgcn_s_sleep(2);
    }
  }
#if __has_builtin(__builtin_amdgcn_s_cluster_barrier)
  __builtin_amdgcn_s_cluster_barrier();   // s_barrier_signal -3 ; s_barrier_wait -3
#endif
  __syncthreads();
  __threadfence();
}

// ---------------- prep kernels ----------------
__global__ void k_cast_f16(_Float16* __restrict__ dst, const float* __restrict__ src, int n) {
  for (int i = blockIdx.x * blockDim.x + threadIdx.x; i < n; i += gridDim.x * blockDim.x)
    dst[i] = (_Float16)src[i];
}

// W1cat[g][576] = [ W_hh1[g][0:512] | W_ih1[g][0:64] ],  Wcov1[g][256] = W_ih1[g][64:320]
__global__ void k_build_w1(_Float16* __restrict__ W1cat, _Float16* __restrict__ Wcov1,
                           const float* __restrict__ W_hh1, const float* __restrict__ W_ih1) {
  const int n1 = 2048 * 576, n2 = 2048 * 256;
  for (int i = blockIdx.x * blockDim.x + threadIdx.x; i < n1 + n2; i += gridDim.x * blockDim.x) {
    if (i < n1) {
      int g = i / 576, k = i - g * 576;
      float v = (k < 512) ? W_hh1[(size_t)g * 512 + k] : W_ih1[(size_t)g * 320 + (k - 512)];
      W1cat[i] = (_Float16)v;
    } else {
      int j = i - n1; int g = j / 256, k = j - g * 256;
      Wcov1[j] = (_Float16)W_ih1[(size_t)g * 320 + 64 + k];
    }
  }
}

__global__ void k_zero(_Float16* __restrict__ hbuf1, _Float16* __restrict__ hbuf2,
                       unsigned* __restrict__ bar) {
  int i0 = blockIdx.x * blockDim.x + threadIdx.x;
  const int n = 2 * BB * H1;   // both ping-pong buffers
  for (int j = i0; j < n; j += gridDim.x * blockDim.x) {
    hbuf1[j] = (_Float16)0.0f;
    hbuf2[j] = (_Float16)0.0f;
  }
  if (i0 < 8) bar[i0] = 0u;
}

// ---------------- persistent dual-LSTM kernel ----------------
__global__ void __launch_bounds__(NTHR)
lstm2_persistent(const _Float16* __restrict__ W1cat,   // [2048][576]
                 const _Float16* __restrict__ Wcov1,   // [2048][256]
                 const _Float16* __restrict__ W2in,    // [2048][768]  (= W_ih2)
                 const _Float16* __restrict__ Whh2,    // [2048][512]
                 const _Float16* __restrict__ inputh,  // [128][365][64]
                 const _Float16* __restrict__ covarh,  // [128][256]
                 const float*    __restrict__ b1v,
                 const float*    __restrict__ b2v,
                 _Float16* __restrict__ hbuf1,         // 2 x [128][512] ping-pong
                 _Float16* __restrict__ hbuf2,         // 2 x [128][512] ping-pong
                 _Float16* __restrict__ hencin,        // [128][768] = [relu(h_enc)|covar]
                 _Float16* __restrict__ hs,            // [90][128][512] relu(h2_t)
                 unsigned* __restrict__ bar)
{
  extern __shared__ _Float16 ldsW[];    // 128 x LDSTR1 halfs, dynamic

  const int w    = blockIdx.x;          // owns hidden cols [w*32, w*32+32)
  const int tid  = threadIdx.x;
  const int lane = tid & 31;
  const int wave = tid >> 5;
  const int m0   = wave * 16;           // batch-row tile (8 waves cover B=128)
  const int hf   = lane >> 4;
  const int rl   = lane & 15;

  // gate column per N-tile: gate q = tn>>1 (i,f,g,o), 16-col block tn&1
  int gc[8];
#pragma unroll
  for (int tn = 0; tn < 8; ++tn)
    gc[tn] = (tn >> 1) * 512 + w * 32 + (tn & 1) * 16 + rl;

  // ---- base1 = b1 + covar @ Wcov1^T (time-invariant part of encoder gates) ----
  v8f base[8];
#pragma unroll
  for (int tn = 0; tn < 8; ++tn) {
    float bv = b1v[gc[tn]];
#pragma unroll
    for (int r = 0; r < 8; ++r) base[tn][r] = bv;
  }
  for (int kk = 0; kk < COV / 32; ++kk) {
    v16h a = load_afrag(covarh, COV, m0, kk * 32, lane);
#pragma unroll
    for (int tn = 0; tn < 8; ++tn)
      base[tn] = wmma32f16(a, load_bfrag_g(Wcov1, COV, gc[tn], kk * 32, lane), base[tn]);
  }

  // ---- stage this WG's 128 encoder gate-columns of W1cat into LDS ----
  // local gate row gl (0..127)  ->  global gate col (gl>>5)*512 + w*32 + (gl&31)
  for (int i = tid; i < 128 * (576 / 8); i += NTHR) {
    int gl = i / 72, ch = i - gl * 72;
    int gg = (gl >> 5) * 512 + w * 32 + (gl & 31);
    *(v8h*)(ldsW + (size_t)gl * LDSTR1 + ch * 8) =
        *(const v8h*)(W1cat + (size_t)gg * 576 + ch * 8);
  }
  __syncthreads();

  float cst[16];                        // cell state, lives in VGPRs across all steps
#pragma unroll
  for (int i = 0; i < 16; ++i) cst[i] = 0.0f;

  grid_sync(bar);                       // hbuf zeros (k_zero) + everyone ready

  // ================= encoder: 365 sequential steps =================
  for (int t = 0; t < TT; ++t) {
    const _Float16* cur = hbuf1 + (size_t)(t & 1) * (BB * H1);
    _Float16*       nxt = hbuf1 + (size_t)((t + 1) & 1) * (BB * H1);
    v8f acc[8];
#pragma unroll
    for (int tn = 0; tn < 8; ++tn) acc[tn] = base[tn];
    for (int kk = 0; kk < 18; ++kk) {   // K = 512 (h) + 64 (x_t)
      v16h a = (kk < 16)
        ? load_afrag(cur, H1, m0, kk * 32, lane)
        : load_afrag(inputh + (size_t)t * INW, TT * INW, m0, (kk - 16) * 32, lane);
#pragma unroll
      for (int tn = 0; tn < 8; ++tn)
        acc[tn] = wmma32f16(a, load_bfrag_lds(ldsW, LDSTR1, tn * 16 + rl, kk * 32, hf),
                            acc[tn]);
    }
    // LSTM pointwise: i/f/g/o are elementwise-aligned across acc tiles
#pragma unroll
    for (int blk = 0; blk < 2; ++blk) {
      v8f iv = acc[blk], fv = acc[2 + blk], gv = acc[4 + blk], ov = acc[6 + blk];
#pragma unroll
      for (int r = 0; r < 8; ++r) {
        float c = sigm(fv[r]) * cst[blk * 8 + r] + sigm(iv[r]) * tanhf(gv[r]);
        float h = sigm(ov[r]) * tanhf(c);
        cst[blk * 8 + r] = c;
        int row = m0 + hf * 8 + r;
        int col = w * 32 + blk * 16 + rl;
        nxt[(size_t)row * H1 + col] = (_Float16)h;
      }
    }
    grid_sync(bar);
  }

  // ---- hencin = [relu(h_enc) | covar] (each WG writes its own slices) ----
  const _Float16* hfin = hbuf1 + (size_t)(TT & 1) * (BB * H1);
  for (int i = tid; i < BB * 32; i += NTHR) {
    int row = i >> 5, cl = i & 31;
    float v = (float)hfin[(size_t)row * H1 + w * 32 + cl];
    hencin[(size_t)row * 768 + w * 32 + cl] = (_Float16)(v > 0.0f ? v : 0.0f);
  }
  for (int i = tid; i < BB * 16; i += NTHR) {
    int row = i >> 4, cl = i & 15;
    hencin[(size_t)row * 768 + H1 + w * 16 + cl] = covarh[(size_t)row * COV + w * 16 + cl];
  }
  grid_sync(bar);

  // ---- base2 = b2 + [relu(h_enc)|covar] @ W_ih2^T (constant all decoder steps) ----
#pragma unroll
  for (int tn = 0; tn < 8; ++tn) {
    float bv = b2v[gc[tn]];
#pragma unroll
    for (int r = 0; r < 8; ++r) base[tn][r] = bv;
  }
  for (int kk = 0; kk < 768 / 32; ++kk) {
    v16h a = load_afrag(hencin, 768, m0, kk * 32, lane);
#pragma unroll
    for (int tn = 0; tn < 8; ++tn)
      base[tn] = wmma32f16(a, load_bfrag_g(W2in, 768, gc[tn], kk * 32, lane), base[tn]);
  }
#pragma unroll
  for (int i = 0; i < 16; ++i) cst[i] = 0.0f;

  // ---- restage LDS with this WG's decoder gate-columns of W_hh2 ----
  // (encoder's last ldsW reads happened before the final grid_sync above)
  for (int i = tid; i < 128 * (512 / 8); i += NTHR) {
    int gl = i >> 6, ch = i & 63;
    int gg = (gl >> 5) * 512 + w * 32 + (gl & 31);
    *(v8h*)(ldsW + (size_t)gl * LDSTR2 + ch * 8) =
        *(const v8h*)(Whh2 + (size_t)gg * 512 + ch * 8);
  }
  __syncthreads();

  // ================= decoder: 90 sequential steps =================
  for (int t = 0; t < OLEN; ++t) {
    const _Float16* cur = hbuf2 + (size_t)(t & 1) * (BB * H2);
    _Float16*       nxt = hbuf2 + (size_t)((t + 1) & 1) * (BB * H2);
    v8f acc[8];
#pragma unroll
    for (int tn = 0; tn < 8; ++tn) acc[tn] = base[tn];
    for (int kk = 0; kk < H2 / 32; ++kk) {
      v16h a = load_afrag(cur, H2, m0, kk * 32, lane);
#pragma unroll
      for (int tn = 0; tn < 8; ++tn)
        acc[tn] = wmma32f16(a, load_bfrag_lds(ldsW, LDSTR2, tn * 16 + rl, kk * 32, hf),
                            acc[tn]);
    }
#pragma unroll
    for (int blk = 0; blk < 2; ++blk) {
      v8f iv = acc[blk], fv = acc[2 + blk], gv = acc[4 + blk], ov = acc[6 + blk];
#pragma unroll
      for (int r = 0; r < 8; ++r) {
        float c = sigm(fv[r]) * cst[blk * 8 + r] + sigm(iv[r]) * tanhf(gv[r]);
        float h = sigm(ov[r]) * tanhf(c);
        cst[blk * 8 + r] = c;
        int row = m0 + hf * 8 + r;
        int col = w * 32 + blk * 16 + rl;
        nxt[(size_t)row * H2 + col] = (_Float16)h;
        float hr = h > 0.0f ? h : 0.0f;
        hs[((size_t)t * BB + row) * H2 + col] = (_Float16)hr;
      }
    }
    grid_sync(bar);
  }
}

// ---------------- output head: out[b,t] = relu(h2_t) . W_fc + b_fc ----------------
__global__ void k_out(const _Float16* __restrict__ hs, const float* __restrict__ wfc,
                      const float* __restrict__ bfc, float* __restrict__ out) {
  int t = blockIdx.x;    // 0..89
  int b = threadIdx.x;   // 0..127
  const _Float16* p = hs + ((size_t)t * BB + b) * H2;
  float acc = bfc[0];
  for (int j = 0; j < H2; ++j) acc += (float)p[j] * wfc[j];
  out[(size_t)b * OLEN + t] = acc;
}

extern "C" void kernel_launch(void* const* d_in, const int* in_sizes, int n_in,
                              void* d_out, int out_size, void* d_ws, size_t ws_size,
                              hipStream_t stream) {
  (void)in_sizes; (void)n_in; (void)out_size; (void)ws_size;
  const float* input = (const float*)d_in[0];   // [128][365][64]
  const float* covar = (const float*)d_in[1];   // [128][256]
  const float* W_ih1 = (const float*)d_in[2];   // [2048][320]
  const float* W_hh1 = (const float*)d_in[3];   // [2048][512]
  const float* b1    = (const float*)d_in[4];   // [2048]
  const float* W_ih2 = (const float*)d_in[5];   // [2048][768]
  const float* W_hh2 = (const float*)d_in[6];   // [2048][512]
  const float* b2    = (const float*)d_in[7];   // [2048]
  const float* W_fc  = (const float*)d_in[8];   // [512]
  const float* b_fc  = (const float*)d_in[9];   // [1]
  float* out = (float*)d_out;

  // carve workspace (f16 arenas, 256B-aligned)
  char* ws = (char*)d_ws;
  size_t off = 0;
  auto carve = [&](size_t nhalf) {
    _Float16* p = (_Float16*)(ws + off);
    off += ((nhalf * sizeof(_Float16)) + 255) & ~(size_t)255;
    return p;
  };
  _Float16* W1cat  = carve((size_t)2048 * 576);
  _Float16* Wcov1  = carve((size_t)2048 * 256);
  _Float16* W2in   = carve((size_t)2048 * 768);
  _Float16* Whh2h  = carve((size_t)2048 * 512);
  _Float16* inputh = carve((size_t)BB * TT * INW);
  _Float16* covarh = carve((size_t)BB * COV);
  _Float16* hencin = carve((size_t)BB * 768);
  _Float16* hbuf1  = carve((size_t)2 * BB * H1);
  _Float16* hbuf2  = carve((size_t)2 * BB * H2);
  _Float16* hs     = carve((size_t)OLEN * BB * H2);
  unsigned* bar    = (unsigned*)(ws + off);

  k_cast_f16<<<512,  NTHR, 0, stream>>>(W2in,  W_ih2, 2048 * 768);
  k_cast_f16<<<512,  NTHR, 0, stream>>>(Whh2h, W_hh2, 2048 * 512);
  k_cast_f16<<<2048, NTHR, 0, stream>>>(inputh, input, BB * TT * INW);
  k_cast_f16<<<64,   NTHR, 0, stream>>>(covarh, covar, BB * COV);
  k_build_w1<<<1024, NTHR, 0, stream>>>(W1cat, Wcov1, W_hh1, W_ih1);
  k_zero<<<256, NTHR, 0, stream>>>(hbuf1, hbuf2, bar);

  lstm2_persistent<<<NWG, NTHR, LDS_BYTES, stream>>>(W1cat, Wcov1, W2in, Whh2h,
                                                     inputh, covarh, b1, b2,
                                                     hbuf1, hbuf2, hencin, hs, bar);

  k_out<<<OLEN, BB, 0, stream>>>(hs, W_fc, b_fc, out);
}
